// GNNRNNForecastModel_81939386073767
// MI455X (gfx1250) — matrix operand order repaired and checked by reading, not compiled, not run on hardware
//
#include <hip/hip_runtime.h>
#include <hip/hip_bf16.h>

// ---------------------------------------------------------------------------
// GNN (GATConv) -> LSTM -> MLP head forecaster for MI455X (gfx1250, wave32).
// LSTM GEMMs (~95% of FLOPs) and the input projection run on the CDNA5 WMMA
// path: v_wmma_f32_16x16x32_f16 (f16 inputs, f32 accumulate). Graph scatter
// parts (segment max/softmax/aggregate) are VALU + float atomics.
// ---------------------------------------------------------------------------

#define BB   2
#define NN   5000
#define EE   40000
#define TT   12
#define FF   16
#define DD   128
#define HH   4
#define HORZ 12
#define CC   32              // DD / HH
#define MM   (BB * NN)       // 10000 LSTM rows
#define G4   (4 * DD)        // 512 gate columns

#define BNTD (BB * NN * TT * DD)   // 15,360,000
#define BNTH (BB * NN * TT * HH)   //    480,000
#define EBTH (EE * BB * TT * HH)   //  3,840,000
#define EBTHC (EBTH * CC)          // 122,880,000
#define ROWS (BB * NN * TT)        //    120,000 rows for x@W_lin

typedef __attribute__((ext_vector_type(16))) _Float16     v16h;
typedef __attribute__((ext_vector_type(8)))  float        v8f;
typedef __attribute__((ext_vector_type(4)))  unsigned int u32x4;
typedef __attribute__((ext_vector_type(4)))  float        f32x4;

union V16H { v16h h; u32x4 u[2]; _Float16 e[16]; };

// ---- device-global scratch (compile-time sizes; independent of ws_size) ----
__device__ float     g_HX  [BNTD];      // GAT-projected features  [B,N,T,D]
__device__ float     g_AGG [BNTD];      // attention aggregation   [B,N,T,D]
__device__ float     g_ASRC[BNTH];      // per-node src scores     [B,N,T,H]
__device__ float     g_ADST[BNTH];      // per-node dst scores
__device__ float     g_AMAX[BNTH];      // segment max
__device__ float     g_DEN [BNTH];      // segment softmax denom
__device__ _Float16  g_GNN16[BNTD];     // LSTM input (f16)        [M,T,D]
__device__ float     g_Hf  [MM * DD];   // LSTM h (f32)
__device__ float     g_Cf  [MM * DD];   // LSTM c (f32)
__device__ _Float16  g_H16 [MM * DD];   // LSTM h (f16, WMMA A operand)
__device__ _Float16  g_WLINT[DD * FF];  // W_lin^T (col-major)  f16
__device__ _Float16  g_WIH [G4 * DD];   // W_ih  f16 row-major [512,128]
__device__ _Float16  g_WHH [G4 * DD];   // W_hh  f16 row-major
__device__ float     g_BIAS[G4];        // b_ih + b_hh
__device__ float     g_S   [HH];        // per-head edge scalar

__device__ __forceinline__ float sigmoidf_(float x) { return 1.f / (1.f + __expf(-x)); }

__device__ __forceinline__ void atomicMaxF(float* addr, float val) {
  unsigned int* ua = reinterpret_cast<unsigned int*>(addr);
  unsigned int old = __float_as_uint(*addr);
  while (__uint_as_float(old) < val) {
    unsigned int assumed = old;
    old = atomicCAS(ua, assumed, __float_as_uint(val));
    if (old == assumed) break;
  }
}

// ---------------------------------------------------------------------------
// 0) weight prep: f32 -> f16 conversions, fused bias, per-head edge scalar
// ---------------------------------------------------------------------------
__global__ __launch_bounds__(256) void k_prep(const float* __restrict__ W_lin,
                                              const float* __restrict__ W_edge,
                                              const float* __restrict__ att_edge,
                                              const float* __restrict__ W_ih,
                                              const float* __restrict__ W_hh,
                                              const float* __restrict__ b_ih,
                                              const float* __restrict__ b_hh) {
  for (int i = blockIdx.x * blockDim.x + threadIdx.x; i < G4 * DD;
       i += gridDim.x * blockDim.x) {
    g_WIH[i] = (_Float16)W_ih[i];
    g_WHH[i] = (_Float16)W_hh[i];
    if (i < DD * FF) {                       // transpose W_lin -> [n][k]
      int n = i / FF, k = i % FF;
      g_WLINT[i] = (_Float16)W_lin[k * DD + n];
    }
    if (i < G4) g_BIAS[i] = b_ih[i] + b_hh[i];
    if (i < HH) {
      float s = 0.f;
      for (int c = 0; c < CC; ++c) s += W_edge[i * CC + c] * att_edge[i * CC + c];
      g_S[i] = s;
    }
  }
}

// ---------------------------------------------------------------------------
// 1) init: zero / -inf the accumulation buffers and LSTM state (every call)
// ---------------------------------------------------------------------------
__global__ __launch_bounds__(256) void k_init() {
  for (int i = blockIdx.x * blockDim.x + threadIdx.x; i < BNTD;
       i += gridDim.x * blockDim.x) {
    g_AGG[i] = 0.f;
    if (i < BNTH) { g_AMAX[i] = -3e38f; g_DEN[i] = 0.f; }
    if (i < MM * DD) { g_Hf[i] = 0.f; g_Cf[i] = 0.f; g_H16[i] = (_Float16)0.f; }
  }
}

// ---------------------------------------------------------------------------
// 2) hx = x @ W_lin  :  [120000,16] x [16,128] via WMMA f16 (K padded to 32)
//    256 threads = 8 waves; each wave owns one 16-row tile, all 8 col tiles.
// ---------------------------------------------------------------------------
__global__ __launch_bounds__(256) void k_lin(const float* __restrict__ x) {
  const int lane = threadIdx.x & 31;
  const int wave = threadIdx.x >> 5;
  const int tile = blockIdx.x * 8 + wave;          // 16-row tile
  if (tile * 16 >= ROWS) return;                    // wave-uniform guard
  const int half = lane >> 4;
  const int row  = tile * 16 + (lane & 15);

  // A tile (16x32): K=0..15 from x (f32->f16), K=16..31 zero padding
  V16H a;
  {
    const f32x4* p = reinterpret_cast<const f32x4*>(x + row * FF + half * 8);
    f32x4 p0 = p[0], p1 = p[1];
    a.e[0] = (_Float16)p0.x; a.e[1] = (_Float16)p0.y;
    a.e[2] = (_Float16)p0.z; a.e[3] = (_Float16)p0.w;
    a.e[4] = (_Float16)p1.x; a.e[5] = (_Float16)p1.y;
    a.e[6] = (_Float16)p1.z; a.e[7] = (_Float16)p1.w;
    #pragma unroll
    for (int j = 8; j < 16; ++j) a.e[j] = (_Float16)0.f;
  }

  #pragma unroll
  for (int ct = 0; ct < 8; ++ct) {
    V16H bm;
    {                                               // elems 0..7: K = 8*half + j (W_lin); 8..15: K pad zeros
      const int n = ct * 16 + (lane & 15);          // column
      const u32x4* p = reinterpret_cast<const u32x4*>(&g_WLINT[n * FF + half * 8]);
      bm.u[0] = p[0];
      #pragma unroll
      for (int j = 8; j < 16; ++j) bm.e[j] = (_Float16)0.f;
    }
    v8f c;
    #pragma unroll
    for (int j = 0; j < 8; ++j) c[j] = 0.f;
    c = __builtin_amdgcn_wmma_f32_16x16x32_f16(false, a.h, false, bm.h,
                                               (short)0, c, false, false);
    const int col   = ct * 16 + (lane & 15);
    const int rbase = tile * 16 + (half ? 8 : 0);
    #pragma unroll
    for (int m = 0; m < 8; ++m) g_HX[(rbase + m) * DD + col] = c[m];
  }
}

// ---------------------------------------------------------------------------
// 3) per-node attention scores a_src / a_dst (dot over C per head)
// ---------------------------------------------------------------------------
__global__ __launch_bounds__(256) void k_attn_node(const float* __restrict__ att_src,
                                                   const float* __restrict__ att_dst) {
  int i = blockIdx.x * blockDim.x + threadIdx.x;
  if (i >= BNTH) return;
  int h = i & 3;
  int base = i >> 2;                  // (b*N+n)*T + t
  const float* hp = &g_HX[base * DD + h * CC];
  float s1 = 0.f, s2 = 0.f;
  for (int c = 0; c < CC; ++c) {
    float v = hp[c];
    s1 += v * att_src[h * CC + c];
    s2 += v * att_dst[h * CC + c];
  }
  g_ASRC[i] = s1; g_ADST[i] = s2;
}

__device__ __forceinline__ float edge_alpha(const int* ei, const float* ew,
                                            int e, int b, int t, int h, int& dnode) {
  int s = ei[e];
  dnode = ei[EE + e];
  float a = g_ASRC[((b * NN + s) * TT + t) * HH + h] +
            g_ADST[((b * NN + dnode) * TT + t) * HH + h] + ew[e] * g_S[h];
  return a > 0.f ? a : 0.2f * a;      // leaky_relu(0.2)
}

// 4) pass 1: segment max over edges sharing dst
__global__ __launch_bounds__(256) void k_edge_max(const int* __restrict__ ei,
                                                  const float* __restrict__ ew) {
  int i = blockIdx.x * blockDim.x + threadIdx.x;
  if (i >= EBTH) return;
  int h = i & 3, i2 = i >> 2;
  int t = i2 % TT, i3 = i2 / TT;
  int b = i3 % BB, e = i3 / BB;
  int dnode;
  float a = edge_alpha(ei, ew, e, b, t, h, dnode);
  atomicMaxF(&g_AMAX[((b * NN + dnode) * TT + t) * HH + h], a);
}

// 5) pass 2: softmax denominator
__global__ __launch_bounds__(256) void k_edge_den(const int* __restrict__ ei,
                                                  const float* __restrict__ ew) {
  int i = blockIdx.x * blockDim.x + threadIdx.x;
  if (i >= EBTH) return;
  int h = i & 3, i2 = i >> 2;
  int t = i2 % TT, i3 = i2 / TT;
  int b = i3 % BB, e = i3 / BB;
  int dnode;
  float a = edge_alpha(ei, ew, e, b, t, h, dnode);
  float ex = __expf(a - g_AMAX[((b * NN + dnode) * TT + t) * HH + h]);
  atomicAdd(&g_DEN[((b * NN + dnode) * TT + t) * HH + h], ex);
}

// 6) pass 3: weighted message aggregation (one atomic per channel element)
__global__ __launch_bounds__(256) void k_edge_msg(const int* __restrict__ ei,
                                                  const float* __restrict__ ew) {
  long long li = (long long)blockIdx.x * blockDim.x + threadIdx.x;
  if (li >= (long long)EBTHC) return;
  int c  = (int)(li & 31);
  long long i1 = li >> 5;
  int h = (int)(i1 & 3);
  long long i2 = i1 >> 2;
  int t = (int)(i2 % TT);
  long long i3 = i2 / TT;
  int b = (int)(i3 % BB);
  int e = (int)(i3 / BB);
  int dnode;
  float a = edge_alpha(ei, ew, e, b, t, h, dnode);
  int nidx = ((b * NN + dnode) * TT + t) * HH + h;
  float ex  = __expf(a - g_AMAX[nidx]);
  float att = ex / (g_DEN[nidx] + 1e-16f);
  int s = ei[e];
  atomicAdd(&g_AGG[((b * NN + dnode) * TT + t) * DD + h * CC + c],
            g_HX[((b * NN + s) * TT + t) * DD + h * CC + c] * att);
}

// 7) gnn = elu(agg + bias) -> f16 LSTM input
__global__ __launch_bounds__(256) void k_elu(const float* __restrict__ gbias) {
  int i = blockIdx.x * blockDim.x + threadIdx.x;
  if (i >= BNTD) return;
  float v = g_AGG[i] + gbias[i & (DD - 1)];
  v = v > 0.f ? v : (__expf(v) - 1.f);
  g_GNN16[i] = (_Float16)v;
}

// ---------------------------------------------------------------------------
// 8) one LSTM time step. Block = 128 threads (4 waves) = one 16-row tile.
//    Wave g computes gate g's [16 x 128] pre-activations: 8 col tiles x
//    (4 K-chunks x {x,h}) = 64 v_wmma_f32_16x16x32_f16 per wave, bias-seeded.
//    Gates staged in LDS, then the pointwise cell update.
// ---------------------------------------------------------------------------
__global__ __launch_bounds__(128) void k_lstm_step(int t) {
  __shared__ float gsm[16 * G4];                    // 32 KB
  const int lane = threadIdx.x & 31;
  const int g    = threadIdx.x >> 5;                // gate i,f,g,o
  const int half = lane >> 4;
  const int ncol = lane & 15;
  const int rowbase = blockIdx.x * 16;
  const int r = rowbase + ncol;                     // this lane's A row

  v8f acc[8];
  #pragma unroll
  for (int ct = 0; ct < 8; ++ct) {
    float bv = g_BIAS[g * DD + ct * 16 + ncol];
    #pragma unroll
    for (int j = 0; j < 8; ++j) acc[ct][j] = bv;
  }

  const _Float16* xrow = &g_GNN16[(size_t)r * (TT * DD) + (size_t)t * DD];
  const _Float16* hrow = &g_H16[(size_t)r * DD];
  const int half8 = half * 8;

  #pragma unroll
  for (int kk = 0; kk < 4; ++kk) {
    const int kb = kk * 32;
    V16H ax, ah;
    ax.u[0] = *reinterpret_cast<const u32x4*>(xrow + kb + half8);
    ax.u[1] = *reinterpret_cast<const u32x4*>(xrow + kb + 16 + half8);
    ah.u[0] = *reinterpret_cast<const u32x4*>(hrow + kb + half8);
    ah.u[1] = *reinterpret_cast<const u32x4*>(hrow + kb + 16 + half8);
    const int kg = kb + half8;
    #pragma unroll
    for (int ct = 0; ct < 8; ++ct) {
      const int n = g * DD + ct * 16 + ncol;        // gate column
      V16H bx, bh;
      const u32x4* pb = reinterpret_cast<const u32x4*>(&g_WIH[n * DD + kg]);
      bx.u[0] = pb[0]; bx.u[1] = pb[2];
      const u32x4* ph = reinterpret_cast<const u32x4*>(&g_WHH[n * DD + kg]);
      bh.u[0] = ph[0]; bh.u[1] = ph[2];
      acc[ct] = __builtin_amdgcn_wmma_f32_16x16x32_f16(false, ax.h, false, bx.h,
                                                       (short)0, acc[ct], false, false);
      acc[ct] = __builtin_amdgcn_wmma_f32_16x16x32_f16(false, ah.h, false, bh.h,
                                                       (short)0, acc[ct], false, false);
    }
  }

  // stage gates: D-layout elem m -> row m (+8 for upper lane group), col=lane%16
  #pragma unroll
  for (int ct = 0; ct < 8; ++ct) {
    const int col = g * DD + ct * 16 + ncol;
    const int rb  = half ? 8 : 0;
    #pragma unroll
    for (int m = 0; m < 8; ++m) gsm[(rb + m) * G4 + col] = acc[ct][m];
  }
  __syncthreads();

  // pointwise cell update: 16 rows x 128 channels, thread = channel
  const int d = threadIdx.x;
  for (int rl = 0; rl < 16; ++rl) {
    float ig = gsm[rl * G4 + d];
    float fg = gsm[rl * G4 + DD + d];
    float gg = gsm[rl * G4 + 2 * DD + d];
    float og = gsm[rl * G4 + 3 * DD + d];
    int R = rowbase + rl;
    float cc = sigmoidf_(fg) * g_Cf[R * DD + d] + sigmoidf_(ig) * tanhf(gg);
    float hh = sigmoidf_(og) * tanhf(cc);
    g_Cf[R * DD + d] = cc;
    g_Hf[R * DD + d] = hh;
    g_H16[R * DD + d] = (_Float16)hh;
  }
}

// ---------------------------------------------------------------------------
// 9) head: LayerNorm -> Linear(128,128) -> ReLU -> Linear(128,12)
//    one row per 128-thread block (tiny GEMMs, VALU is fine)
// ---------------------------------------------------------------------------
__global__ __launch_bounds__(128) void k_head(const float* __restrict__ ln_g,
                                              const float* __restrict__ ln_b,
                                              const float* __restrict__ W1,
                                              const float* __restrict__ b1,
                                              const float* __restrict__ W2,
                                              const float* __restrict__ b2,
                                              float* __restrict__ out) {
  __shared__ float sred[128];
  __shared__ float sln[128];
  __shared__ float sz[128];
  const int row = blockIdx.x;
  const int tid = threadIdx.x;
  float v = g_Hf[row * DD + tid];

  sred[tid] = v; __syncthreads();
  #pragma unroll
  for (int s = 64; s > 0; s >>= 1) { if (tid < s) sred[tid] += sred[tid + s]; __syncthreads(); }
  float mu = sred[0] * (1.f / DD); __syncthreads();
  float dv = v - mu;
  sred[tid] = dv * dv; __syncthreads();
  #pragma unroll
  for (int s = 64; s > 0; s >>= 1) { if (tid < s) sred[tid] += sred[tid + s]; __syncthreads(); }
  float var = sred[0] * (1.f / DD); __syncthreads();

  sln[tid] = dv * rsqrtf(var + 1e-5f) * ln_g[tid] + ln_b[tid];
  __syncthreads();

  float a1 = b1[tid];
  for (int k = 0; k < DD; ++k) a1 += sln[k] * W1[k * DD + tid];
  sz[tid] = fmaxf(a1, 0.f);
  __syncthreads();

  if (tid < HORZ) {
    float a2 = b2[tid];
    for (int k = 0; k < DD; ++k) a2 += sz[k] * W2[k * HORZ + tid];
    out[row * HORZ + tid] = a2;
  }
}

// ---------------------------------------------------------------------------
extern "C" void kernel_launch(void* const* d_in, const int* in_sizes, int n_in,
                              void* d_out, int out_size, void* d_ws, size_t ws_size,
                              hipStream_t stream) {
  (void)in_sizes; (void)n_in; (void)out_size; (void)d_ws; (void)ws_size;
  const float* x        = (const float*)d_in[0];
  const int*   ei       = (const int*)  d_in[1];
  const float* ew       = (const float*)d_in[2];
  const float* W_lin    = (const float*)d_in[3];
  const float* att_src  = (const float*)d_in[4];
  const float* att_dst  = (const float*)d_in[5];
  const float* W_edge   = (const float*)d_in[6];
  const float* att_edge = (const float*)d_in[7];
  const float* gat_bias = (const float*)d_in[8];
  const float* W_ih     = (const float*)d_in[9];
  const float* W_hh     = (const float*)d_in[10];
  const float* b_ih     = (const float*)d_in[11];
  const float* b_hh     = (const float*)d_in[12];
  const float* ln_g     = (const float*)d_in[13];
  const float* ln_b     = (const float*)d_in[14];
  const float* W1       = (const float*)d_in[15];
  const float* b1       = (const float*)d_in[16];
  const float* W2       = (const float*)d_in[17];
  const float* b2       = (const float*)d_in[18];
  float* out = (float*)d_out;

  k_prep<<<(G4 * DD + 255) / 256, 256, 0, stream>>>(W_lin, W_edge, att_edge,
                                                    W_ih, W_hh, b_ih, b_hh);
  k_init<<<BNTD / 256, 256, 0, stream>>>();
  k_lin<<<(ROWS / 16 + 7) / 8, 256, 0, stream>>>(x);
  k_attn_node<<<BNTH / 256, 256, 0, stream>>>(att_src, att_dst);
  k_edge_max<<<EBTH / 256, 256, 0, stream>>>(ei, ew);
  k_edge_den<<<EBTH / 256, 256, 0, stream>>>(ei, ew);
  k_edge_msg<<<EBTHC / 256, 256, 0, stream>>>(ei, ew);
  k_elu<<<BNTD / 256, 256, 0, stream>>>(gat_bias);
  for (int t = 0; t < TT; ++t)
    k_lstm_step<<<MM / 16, 128, 0, stream>>>(t);
  k_head<<<MM, 128, 0, stream>>>(ln_g, ln_b, W1, b1, W2, b2, out);
}
